// MultiHeadAttention_24584392802661
// MI455X (gfx1250) — matrix operand order repaired and checked by reading, not compiled, not run on hardware
//
#include <hip/hip_runtime.h>
#include <hip/hip_bf16.h>

typedef __bf16 bf16;
typedef __attribute__((ext_vector_type(8)))  __bf16 v8bf;
typedef __attribute__((ext_vector_type(16))) __bf16 v16bf;
typedef __attribute__((ext_vector_type(8)))  float  v8f;

#define WMMA_BF16(a, b, c)                                              \
  __builtin_amdgcn_wmma_f32_16x16x32_bf16(false, (a), false, (b),       \
                                          (short)0, (c), false, false)

// ---------------------------------------------------------------------------
// Async global -> LDS copies (CDNA5 GLOBAL_LOAD_ASYNC_TO_LDS_B128, ASYNCcnt).
// Guarded: falls back to load+ds_store if the toolchain lacks the builtins.
// Builtin signature (from hipcc diagnostic): param0 = v4i (vector_size 16)
// pointer in AS(1) (global), param1 = LDS side, then i32 offset, i32 cpol.
// ---------------------------------------------------------------------------
#if defined(__has_builtin)
#if __has_builtin(__builtin_amdgcn_global_load_async_to_lds_b128) && \
    __has_builtin(__builtin_amdgcn_s_wait_asynccnt)
#define USE_ASYNC_LDS 1
#endif
#endif
#ifndef USE_ASYNC_LDS
#define USE_ASYNC_LDS 0
#endif

typedef int vi4_t __attribute__((vector_size(16)));
typedef __attribute__((address_space(1))) vi4_t* gvi4_p;
typedef __attribute__((address_space(3))) vi4_t* lvi4_p;

__device__ inline void async_copy16(bf16* lds_dst, const bf16* gsrc) {
#if USE_ASYNC_LDS
  __builtin_amdgcn_global_load_async_to_lds_b128((gvi4_p)gsrc, (lvi4_p)lds_dst,
                                                 0, 0);
#else
  *(v8bf*)lds_dst = *(const v8bf*)gsrc;
#endif
}

__device__ inline void async_wait_all() {
#if USE_ASYNC_LDS
  __builtin_amdgcn_s_wait_asynccnt(0);
#endif
}

// ---------------------------------------------------------------------------
// LDS fragment loaders (CDNA5 wave32 WMMA 16x16x32 bf16 layouts, ISA 7.12.2)
// A (16x32, row-major in LDS, row stride LD elements):
//   lane: m = lane&15, kbase = (lane>>4)*8 -> chunks [kbase,kbase+8) and
//   [kbase+16,kbase+24)  => two contiguous 16B ds loads.
// B (32x16): staged "transposed" [n][k] so lane (col n = lane&15) reads K
//   chunks [kbase,kbase+8) and [kbase+16,kbase+24), kbase=(lane>>4)*8 => two 16B ds loads.
// ---------------------------------------------------------------------------
template <int LD>
__device__ inline v16bf load_frag_A(const bf16* base, int lane) {
  int m  = lane & 15;
  int c0 = (lane >> 4) * 8;
  const bf16* rp = base + m * LD + c0;
  v8bf lo = *(const v8bf*)(rp);
  v8bf hi = *(const v8bf*)(rp + 16);
  return __builtin_shufflevector(lo, hi, 0, 1, 2, 3, 4, 5, 6, 7,
                                 8, 9, 10, 11, 12, 13, 14, 15);
}

template <int LD>
__device__ inline v16bf load_frag_B(const bf16* base, int lane) {
  int n  = lane & 15;
  int k0 = (lane >> 4) * 8;
  const bf16* rp = base + n * LD + k0;
  v8bf lo = *(const v8bf*)(rp);
  v8bf hi = *(const v8bf*)(rp + 16);
  return __builtin_shufflevector(lo, hi, 0, 1, 2, 3, 4, 5, 6, 7,
                                 8, 9, 10, 11, 12, 13, 14, 15);
}

// ---------------------------------------------------------------------------
// fp32 -> bf16 bulk convert
// ---------------------------------------------------------------------------
__global__ void f32_to_bf16_kernel(const float* __restrict__ in,
                                   bf16* __restrict__ out, long n) {
  long i = (long)blockIdx.x * blockDim.x + threadIdx.x;
  if (i < n) out[i] = (bf16)in[i];
}

// ---------------------------------------------------------------------------
// GEMM: C[M,N] = A[M,K](bf16) * B[K,N](bf16) + bias[N], fp32 accumulate.
// Block tile 128x128, K-step 32. 8 waves as 4(M) x 2(N); each wave: 32x64
// output = 2x4 WMMA accumulators (8 WMMAs per K-step per wave).
// A tile staged via async global->LDS; B tile transposed through VGPRs.
// ---------------------------------------------------------------------------
template <bool OUT_BF16>
__global__ __launch_bounds__(256) void gemm_bias_wmma(
    const bf16* __restrict__ A, const bf16* __restrict__ Bw,
    const float* __restrict__ bias, void* __restrict__ Cout,
    int M, int N, int K) {
  constexpr int BM = 128, BN = 128, BK = 32;
  __shared__ __align__(16) bf16 As[BM][BK];  // 8 KB
  __shared__ __align__(16) bf16 Bs[BN][BK];  // 8 KB, [n][k]

  int tid  = threadIdx.x;
  int lane = tid & 31;
  int wave = tid >> 5;
  int wm = wave >> 1;  // 0..3  (32 rows each)
  int wn = wave & 1;   // 0..1  (64 cols each)
  int m0 = blockIdx.y * BM;
  int n0 = blockIdx.x * BN;

  // Staging thread mappings.
  int ar = tid >> 1;            // A: row 0..127
  int ac = (tid & 1) * 16;      // A: col 0 or 16
  int btk = tid & 31;           // B: k within tile
  int btn = tid >> 5;           // B: 16-col group 0..7

  v8f acc[2][4] = {};

  for (int k0 = 0; k0 < K; k0 += BK) {
    // Stage A tile: async direct-to-LDS (2 x 16B per thread).
    {
      const bf16* gp = A + (size_t)(m0 + ar) * K + k0 + ac;
      async_copy16(&As[ar][ac], gp);
      async_copy16(&As[ar][ac + 8], gp + 8);
      if (k0 + BK < K) __builtin_prefetch(gp + BK, 0, 3);
    }
    // Stage B tile transposed into [n][k] (16 cols per thread).
    {
      const bf16* gp = Bw + (size_t)(k0 + btk) * N + n0 + btn * 16;
      v8bf v0 = *(const v8bf*)(gp);
      v8bf v1 = *(const v8bf*)(gp + 8);
#pragma unroll
      for (int j = 0; j < 8; ++j) {
        Bs[btn * 16 + j][btk]     = v0[j];
        Bs[btn * 16 + 8 + j][btk] = v1[j];
      }
      if (k0 + BK < K) __builtin_prefetch(gp + (size_t)BK * N, 0, 3);
    }
    async_wait_all();
    __syncthreads();
#pragma unroll
    for (int mt = 0; mt < 2; ++mt) {
      v16bf a = load_frag_A<BK>(&As[wm * 32 + mt * 16][0], lane);
#pragma unroll
      for (int nt = 0; nt < 4; ++nt) {
        v16bf b = load_frag_B<BK>(&Bs[wn * 64 + nt * 16][0], lane);
        acc[mt][nt] = WMMA_BF16(a, b, acc[mt][nt]);
      }
    }
    __syncthreads();
  }

  // Epilogue: C/D layout -> row = base + r + (lane>>4)*8, col = base + lane&15
  int colin  = lane & 15;
  int rowoff = (lane >> 4) * 8;
#pragma unroll
  for (int mt = 0; mt < 2; ++mt) {
#pragma unroll
    for (int nt = 0; nt < 4; ++nt) {
      int col  = n0 + wn * 64 + nt * 16 + colin;
      float bv = bias[col];
#pragma unroll
      for (int r = 0; r < 8; ++r) {
        int row = m0 + wm * 32 + mt * 16 + rowoff + r;
        float v = acc[mt][nt][r] + bv;
        if (OUT_BF16)
          ((bf16*)Cout)[(size_t)row * N + col] = (bf16)v;
        else
          ((float*)Cout)[(size_t)row * N + col] = v;
      }
    }
  }
}

// ---------------------------------------------------------------------------
// Split qkv (B, L*3E) bf16 into dense Q/K/V (B*H, L, 64) bf16 per the
// reference's position-mixing reshape: per (b,h) the chunk at
// b*L*3072 + h*L*192 holds rows of 192 = [q(64) | k(64) | v(64)].
// ---------------------------------------------------------------------------
__global__ void split_qkv_kernel(const bf16* __restrict__ qkv,
                                 bf16* __restrict__ Qb, bf16* __restrict__ Kb,
                                 bf16* __restrict__ Vb) {
  int idx = blockIdx.x * blockDim.x + threadIdx.x;  // (bh, l, d8): 32*2048*8
  int d8 = idx & 7;
  int l  = (idx >> 3) & 2047;
  int bh = idx >> 14;  // 0..31
  int b = bh >> 4, h = bh & 15;
  size_t src = (size_t)b * (2048 * 3072) + (size_t)h * (2048 * 192) +
               (size_t)l * 192 + d8 * 8;
  size_t dst = ((size_t)bh * 2048 + l) * 64 + d8 * 8;
  *(v8bf*)(Qb + dst) = *(const v8bf*)(qkv + src);
  *(v8bf*)(Kb + dst) = *(const v8bf*)(qkv + src + 64);
  *(v8bf*)(Vb + dst) = *(const v8bf*)(qkv + src + 128);
}

// ---------------------------------------------------------------------------
// Flash attention, one 64-row q-block of one (b,h) per workgroup.
// 8 waves in 4(M) x 2(N). S and O both 64x64 -> each wave owns 16 rows x 32
// cols. Q.K^T and P.V via WMMA bf16; online softmax in fp32.
// Q/K tiles staged via async global->LDS; V transposed through VGPRs.
// Output written directly into (B*L, E) bf16 at column h*64+d.
// ---------------------------------------------------------------------------
__global__ __launch_bounds__(256) void flash_attn_wmma(
    const bf16* __restrict__ Qb, const bf16* __restrict__ Kb,
    const bf16* __restrict__ Vb, bf16* __restrict__ attn) {
  constexpr int L = 2048, DH = 64, E = 1024;
  __shared__ __align__(16) bf16 Qs[64][DH];    // 8 KB
  __shared__ __align__(16) bf16 Ks[64][DH];    // 8 KB  [key][d]
  __shared__ __align__(16) bf16 Vts[DH][64];   // 8 KB  [d][key]
  __shared__ __align__(16) float Ss[64][64];   // 16 KB
  __shared__ __align__(16) bf16 Ps[64][64];    // 8 KB
  __shared__ float mrow[64], lrow[64], crow[64];

  int tid = threadIdx.x, lane = tid & 31, wave = tid >> 5;
  int wm = wave >> 1, wn = wave & 1;
  int bh = blockIdx.x;  // 0..31
  int qb = blockIdx.y;  // 0..31
  int b = bh >> 4, h = bh & 15;

  const bf16* Qp = Qb + ((size_t)bh * L + qb * 64) * DH;
  const bf16* Kp = Kb + (size_t)bh * L * DH;
  const bf16* Vp = Vb + (size_t)bh * L * DH;

  int sr = tid >> 2;           // staging row 0..63
  int sc = (tid & 3) * 16;     // staging col 0/16/32/48

  // Load Q block (async; completion covered by the wait inside the loop).
  async_copy16(&Qs[sr][sc], Qp + sr * DH + sc);
  async_copy16(&Qs[sr][sc + 8], Qp + sr * DH + sc + 8);

  if (tid < 64) {
    mrow[tid] = -1e30f;
    lrow[tid] = 0.0f;
  }

  v8f o[2] = {};
  const float scale = 0.125f;  // 1/sqrt(64)

  for (int kb = 0; kb < L / 64; ++kb) {
    __syncthreads();  // previous iteration done reading Ks/Vts/Ps
    // K block: async straight copy. V block: transpose through VGPRs.
    {
      const bf16* kp = Kp + (size_t)(kb * 64 + sr) * DH + sc;
      async_copy16(&Ks[sr][sc], kp);
      async_copy16(&Ks[sr][sc + 8], kp + 8);
      const bf16* vp = Vp + (size_t)(kb * 64 + sr) * DH + sc;
      v8bf v0 = *(const v8bf*)(vp);
      v8bf v1 = *(const v8bf*)(vp + 8);
#pragma unroll
      for (int j = 0; j < 8; ++j) {
        Vts[sc + j][sr]     = v0[j];
        Vts[sc + 8 + j][sr] = v1[j];
      }
    }
    async_wait_all();
    __syncthreads();

    // S = scale * Q @ K^T   (B-fragment = contiguous K rows of Ks)
#pragma unroll
    for (int nt = 0; nt < 2; ++nt) {
      v8f s = {};
#pragma unroll
      for (int kk = 0; kk < 2; ++kk) {
        v16bf a = load_frag_A<DH>(&Qs[wm * 16][kk * 32], lane);
        v16bf bb = load_frag_B<DH>(&Ks[wn * 32 + nt * 16][kk * 32], lane);
        s = WMMA_BF16(a, bb, s);
      }
      int col   = wn * 32 + nt * 16 + (lane & 15);
      int rbase = wm * 16 + (lane >> 4) * 8;
#pragma unroll
      for (int r = 0; r < 8; ++r) Ss[rbase + r][col] = s[r] * scale;
    }
    __syncthreads();

    // Online softmax: one q-row per thread.
    if (tid < 64) {
      float mo = mrow[tid];
      float bm = mo;
#pragma unroll 8
      for (int c = 0; c < 64; ++c) bm = fmaxf(bm, Ss[tid][c]);
      float corr = __expf(mo - bm);
      float sum  = 0.0f;
#pragma unroll 8
      for (int c = 0; c < 64; ++c) {
        float p = __expf(Ss[tid][c] - bm);
        Ps[tid][c] = (bf16)p;
        sum += p;
      }
      mrow[tid] = bm;
      lrow[tid] = lrow[tid] * corr + sum;
      crow[tid] = corr;
    }
    __syncthreads();

    // Rescale O accumulators by this block's correction.
    {
      int rbase = wm * 16 + (lane >> 4) * 8;
#pragma unroll
      for (int r = 0; r < 8; ++r) {
        float c = crow[rbase + r];
        o[0][r] *= c;
        o[1][r] *= c;
      }
    }
    // O += P @ V   (B-fragment = contiguous rows of Vts)
#pragma unroll
    for (int nt = 0; nt < 2; ++nt) {
#pragma unroll
      for (int kk = 0; kk < 2; ++kk) {
        v16bf a = load_frag_A<64>(&Ps[wm * 16][kk * 32], lane);
        v16bf bb = load_frag_B<64>(&Vts[wn * 32 + nt * 16][kk * 32], lane);
        o[nt] = WMMA_BF16(a, bb, o[nt]);
      }
    }
  }
  __syncthreads();

  // Normalize and scatter into (B*L, E) at column h*64 + d.
  {
    int dcol  = wn * 32 + (lane & 15);
    int rbase = wm * 16 + (lane >> 4) * 8;
#pragma unroll
    for (int nt = 0; nt < 2; ++nt) {
      int d = dcol + nt * 16;
#pragma unroll
      for (int r = 0; r < 8; ++r) {
        int row   = rbase + r;
        float inv = 1.0f / lrow[row];
        int grow  = qb * 64 + row;
        attn[((size_t)(b * L + grow)) * E + h * 64 + d] = (bf16)(o[nt][r] * inv);
      }
    }
  }
}

// ---------------------------------------------------------------------------
// Host launcher
// ---------------------------------------------------------------------------
extern "C" void kernel_launch(void* const* d_in, const int* in_sizes, int n_in,
                              void* d_out, int out_size, void* d_ws,
                              size_t ws_size, hipStream_t stream) {
  (void)in_sizes; (void)n_in; (void)out_size; (void)ws_size;
  constexpr int B = 2, L = 2048, E = 1024, H = 16, DH = 64;
  constexpr int M = B * L;       // 4096
  constexpr int N3 = 3 * E;      // 3072

  const float* x     = (const float*)d_in[0];
  const float* W_qkv = (const float*)d_in[1];
  const float* b_qkv = (const float*)d_in[2];
  const float* W_out = (const float*)d_in[3];
  const float* b_out = (const float*)d_in[4];
  float* out = (float*)d_out;

  // Workspace carve-up (bf16 buffers, 256B aligned).
  char* ws = (char*)d_ws;
  auto alloc = [&](size_t bytes) {
    char* p = ws;
    ws += (bytes + 255) & ~size_t(255);
    return p;
  };
  bf16* xb    = (bf16*)alloc((size_t)M * E * 2);        // 8 MB
  bf16* Wqb   = (bf16*)alloc((size_t)E * N3 * 2);       // 6 MB
  bf16* Wob   = (bf16*)alloc((size_t)E * E * 2);        // 2 MB
  bf16* qkvb  = (bf16*)alloc((size_t)M * N3 * 2);       // 24 MB
  bf16* Qbuf  = (bf16*)alloc((size_t)B * H * L * DH * 2);
  bf16* Kbuf  = (bf16*)alloc((size_t)B * H * L * DH * 2);
  bf16* Vbuf  = (bf16*)alloc((size_t)B * H * L * DH * 2);
  bf16* attnb = (bf16*)alloc((size_t)M * E * 2);        // 8 MB

  // 1) fp32 -> bf16 conversions.
  {
    long n = (long)M * E;
    f32_to_bf16_kernel<<<(n + 255) / 256, 256, 0, stream>>>(x, xb, n);
  }
  {
    long n = (long)E * N3;
    f32_to_bf16_kernel<<<(n + 255) / 256, 256, 0, stream>>>(W_qkv, Wqb, n);
  }
  {
    long n = (long)E * E;
    f32_to_bf16_kernel<<<(n + 255) / 256, 256, 0, stream>>>(W_out, Wob, n);
  }

  // 2) QKV projection: qkvb = xb @ Wqb + b_qkv (bf16 out).
  {
    dim3 grid(N3 / 128, M / 128);
    gemm_bias_wmma<true><<<grid, 256, 0, stream>>>(xb, Wqb, b_qkv, qkvb,
                                                   M, N3, E);
  }

  // 3) Repack into dense per-head Q/K/V.
  {
    int threads = B * H * L * (DH / 8);  // 524288
    split_qkv_kernel<<<threads / 256, 256, 0, stream>>>(qkvb, Qbuf, Kbuf,
                                                        Vbuf);
  }

  // 4) Flash attention -> attnb (B*L, E) bf16.
  {
    dim3 grid(B * H, L / 64);
    flash_attn_wmma<<<grid, 256, 0, stream>>>(Qbuf, Kbuf, Vbuf, attnb);
  }

  // 5) Output projection: out = attnb @ Wob + b_out (fp32 out).
  {
    dim3 grid(E / 128, M / 128);
    gemm_bias_wmma<false><<<grid, 256, 0, stream>>>(attnb, Wob, b_out, out,
                                                    M, E, E);
  }
}
